// Multihead_Attention_Main_78262894068482
// MI455X (gfx1250) — matrix operand order repaired and checked by reading, not compiled, not run on hardware
//
#include <hip/hip_runtime.h>

// ---------------------------------------------------------------------------
// Types for CDNA5 WMMA (wave32, 16x16x32 bf16 -> f32)
// ---------------------------------------------------------------------------
typedef __attribute__((ext_vector_type(16))) __bf16       bf16x16;
typedef __attribute__((ext_vector_type(8)))  float        f32x8;
typedef __attribute__((ext_vector_type(4)))  float        f32x4;
typedef __attribute__((ext_vector_type(4)))  unsigned int u32x4;
typedef __attribute__((ext_vector_type(8)))  unsigned int u32x8;

#define WMMA_BF16(a, b, c) \
  __builtin_amdgcn_wmma_f32_16x16x32_bf16(false, (a), false, (b), (short)0, (c), false, false)

static __device__ __forceinline__ unsigned short f2bf(float f) {
  unsigned int u = __builtin_bit_cast(unsigned int, f);
  return (unsigned short)((u + 0x7fffu + ((u >> 16) & 1u)) >> 16);  // RNE
}

// A-fragment (16x32 bf16): lanes 0-15 row M=lane (K=k0+0..7 / +16..23),
// lanes 16-31 row M=lane-16 (K=k0+8..15 / +24..31). Two 16B loads per lane.
static __device__ __forceinline__ bf16x16
frag_a(const unsigned short* __restrict__ base, int ld, int m0, int k0) {
  const int lane = threadIdx.x & 31;
  const int hf   = lane >> 4;
  const int r    = lane & 15;
  const unsigned short* p = base + (size_t)(m0 + r) * ld + k0 + hf * 8;
  u32x4 lo = *(const u32x4*)(p);
  u32x4 hi = *(const u32x4*)(p + 16);
  u32x8 v  = {lo.x, lo.y, lo.z, lo.w, hi.x, hi.y, hi.z, hi.w};
  return __builtin_bit_cast(bf16x16, v);
}

// B-fragment (32x16 bf16): lanes 0-15 col N=lane K=k0+0..7 / +16..23, lanes 16-31
// col N=lane-16 K=k0+8..15 / +24..31. Column N contiguous along K in memory
// (row-major [N, ld]) -> two 16B loads per lane.
static __device__ __forceinline__ bf16x16
frag_b(const unsigned short* __restrict__ base, int ld, int n0, int k0) {
  const int lane = threadIdx.x & 31;
  const int hf   = lane >> 4;
  const int c    = lane & 15;
  const unsigned short* p = base + (size_t)(n0 + c) * ld + k0 + hf * 8;
  u32x4 lo = *(const u32x4*)(p);
  u32x4 hi = *(const u32x4*)(p + 16);
  u32x8 v  = {lo.x, lo.y, lo.z, lo.w, hi.x, hi.y, hi.z, hi.w};
  return __builtin_bit_cast(bf16x16, v);
}

// ---------------------------------------------------------------------------
// Fused fp32 -> bf16 for x + 4 weight matrices, 8 elements per thread.
// Index space: [0, NX/8) -> x; then 4 regions of NW/8 for Wq/Wk/Wv/Wo.
// ---------------------------------------------------------------------------
struct CvtArgs {
  const float* src[5];
  unsigned int* dst[5];
};

__global__ __launch_bounds__(256) void cvt_all_kernel(CvtArgs args, int nx8, int nw8) {
  int i = blockIdx.x * blockDim.x + threadIdx.x;
  int region, off;
  if (i < nx8) {
    region = 0; off = i;
  } else {
    int j = i - nx8;
    region = 1 + j / nw8;
    off = j % nw8;
    if (region > 4) return;
  }
  const f32x4* p = (const f32x4*)args.src[region] + (size_t)off * 2;
  f32x4 a = p[0], b = p[1];
  u32x4 o;
  o.x = (unsigned)f2bf(a.x) | ((unsigned)f2bf(a.y) << 16);
  o.y = (unsigned)f2bf(a.z) | ((unsigned)f2bf(a.w) << 16);
  o.z = (unsigned)f2bf(b.x) | ((unsigned)f2bf(b.y) << 16);
  o.w = (unsigned)f2bf(b.z) | ((unsigned)f2bf(b.w) << 16);
  ((u32x4*)args.dst[region])[off] = o;
}

// ---------------------------------------------------------------------------
// Fused QKV projection: blockIdx.z selects {Wq->Q, Wk->K, Wv->V^T}.
// Y = X @ W^T. A=[4096,1024] bf16, W=[1024,1024] bf16 [n,k].
// Wave tile 32x64 (8 accumulators); 8 waves/block -> block tile 256x64.
// Q/K written [b,h,t,hd]; V written [b,h,hd,t].
// ---------------------------------------------------------------------------
__global__ __launch_bounds__(256) void qkv_proj_kernel(
    const unsigned short* __restrict__ A,
    const unsigned short* __restrict__ Wq, const unsigned short* __restrict__ Wk,
    const unsigned short* __restrict__ Wv,
    unsigned short* __restrict__ Qo, unsigned short* __restrict__ Ko,
    unsigned short* __restrict__ Vo) {
  const int z = blockIdx.z;
  const unsigned short* W = (z == 0) ? Wq : (z == 1) ? Wk : Wv;
  unsigned short* Out = (z == 0) ? Qo : (z == 1) ? Ko : Vo;
  const int transposeV = (z == 2);

  const int wave = threadIdx.x >> 5;
  const int lane = threadIdx.x & 31;
  const int m0 = blockIdx.x * 256 + wave * 32;
  const int n0 = blockIdx.y * 64;

  f32x8 acc[2][4] = {};
  for (int k = 0; k < 1024; k += 32) {
    if (k + 32 < 1024) {  // WGP-scope prefetch of next k-step
      __builtin_prefetch(A + (size_t)(m0 + lane) * 1024 + k + 32, 0, 3);
      __builtin_prefetch(W + (size_t)(n0 + lane) * 1024 + k + 32, 0, 3);
      __builtin_prefetch(W + (size_t)(n0 + 32 + lane) * 1024 + k + 32, 0, 3);
    }
    bf16x16 a0 = frag_a(A, 1024, m0, k);
    bf16x16 a1 = frag_a(A, 1024, m0 + 16, k);
#pragma unroll
    for (int j = 0; j < 4; ++j) {
      bf16x16 b = frag_b(W, 1024, n0 + 16 * j, k);
      acc[0][j] = WMMA_BF16(a0, b, acc[0][j]);
      acc[1][j] = WMMA_BF16(a1, b, acc[1][j]);
    }
  }
  const int hf = lane >> 4, c = lane & 15;
#pragma unroll
  for (int i = 0; i < 2; ++i)
#pragma unroll
    for (int j = 0; j < 4; ++j)
#pragma unroll
      for (int r = 0; r < 8; ++r) {
        int gm = m0 + 16 * i + r + hf * 8;     // global row in [0,4096)
        int gn = n0 + 16 * j + c;              // global col in [0,1024)
        int bb = gm >> 11, tpos = gm & 2047;   // batch, time
        int head = gn >> 6, hd = gn & 63;      // head, head-dim
        size_t idx = transposeV
            ? ((size_t)((bb * 16 + head) * 64 + hd) * 2048 + tpos)
            : ((size_t)((bb * 16 + head) * 2048 + tpos) * 64 + hd);
        Out[idx] = f2bf(acc[i][j][r]);
      }
}

// ---------------------------------------------------------------------------
// Flash-style attention. One wave owns a 16-row q tile of one (b,h);
// streams K/V in 64-wide kv blocks with online softmax.
// Q,K: [32,2048,64] bf16;  Vt: [32,64,2048] bf16;  Ctx: [2,2048,1024] bf16
// ---------------------------------------------------------------------------
__global__ __launch_bounds__(256) void attn_kernel(
    const unsigned short* __restrict__ Q, const unsigned short* __restrict__ K,
    const unsigned short* __restrict__ Vt, unsigned short* __restrict__ Ctx) {
  __shared__ unsigned short P[8][16][64];  // per-wave P-tile bounce (16KB)

  const int wave = threadIdx.x >> 5;
  const int lane = threadIdx.x & 31;
  const int tile = blockIdx.x * 8 + wave;  // 4096 q-tiles total
  const int bh = tile >> 7;                // 128 tiles per (b,h)
  const int q0 = (tile & 127) * 16;

  const unsigned short* Qb = Q  + (size_t)bh * 2048 * 64;
  const unsigned short* Kb = K  + (size_t)bh * 2048 * 64;
  const unsigned short* Vb = Vt + (size_t)bh * 64 * 2048;
  unsigned short* Pw = &P[wave][0][0];

  // Q fragments held in registers for the whole kv sweep (hd = 64 -> 2 frags)
  bf16x16 qa0 = frag_a(Qb, 64, q0, 0);
  bf16x16 qa1 = frag_a(Qb, 64, q0, 32);

  f32x8 cacc[4] = {};
  float m_i[8], l_i[8];
#pragma unroll
  for (int r = 0; r < 8; ++r) { m_i[r] = -1e30f; l_i[r] = 0.0f; }
  const float scale = 0.125f;  // 1/sqrt(64)
  const int hf = lane >> 4, c = lane & 15;

  for (int n0 = 0; n0 < 2048; n0 += 64) {
    if (n0 + 64 < 2048) {  // WGP-scope prefetch of next kv block
      __builtin_prefetch(Kb + (size_t)(n0 + 64 + lane) * 64, 0, 3);
      __builtin_prefetch(Kb + (size_t)(n0 + 96 + lane) * 64, 0, 3);
      __builtin_prefetch(Vb + (size_t)lane * 2048 + n0 + 64, 0, 3);
      __builtin_prefetch(Vb + (size_t)(lane + 32) * 2048 + n0 + 64, 0, 3);
    }
    // S = Q.K^T for kv block [n0, n0+64): four 16x16 accumulators
    f32x8 s[4] = {};
#pragma unroll
    for (int t4 = 0; t4 < 4; ++t4) {
      s[t4] = WMMA_BF16(qa0, frag_b(Kb, 64, n0 + 16 * t4, 0), s[t4]);
      s[t4] = WMMA_BF16(qa1, frag_b(Kb, 64, n0 + 16 * t4, 32), s[t4]);
    }

    // Online softmax; stats per accumulator-VGPR row
    // (lanes 0-15: rows 0-7, lanes 16-31: rows 8-15); 16-lane butterflies.
    float corr[8];
#pragma unroll
    for (int r = 0; r < 8; ++r) {
      float v0 = s[0][r] * scale, v1 = s[1][r] * scale;
      float v2 = s[2][r] * scale, v3 = s[3][r] * scale;
      float mx = fmaxf(fmaxf(v0, v1), fmaxf(v2, v3));
      mx = fmaxf(mx, __shfl_xor(mx, 1, 32));
      mx = fmaxf(mx, __shfl_xor(mx, 2, 32));
      mx = fmaxf(mx, __shfl_xor(mx, 4, 32));
      mx = fmaxf(mx, __shfl_xor(mx, 8, 32));
      float mn = fmaxf(m_i[r], mx);
      float cr = __expf(m_i[r] - mn);
      float p0 = __expf(v0 - mn), p1 = __expf(v1 - mn);
      float p2 = __expf(v2 - mn), p3 = __expf(v3 - mn);
      float rs = (p0 + p1) + (p2 + p3);
      rs += __shfl_xor(rs, 1, 32);
      rs += __shfl_xor(rs, 2, 32);
      rs += __shfl_xor(rs, 4, 32);
      rs += __shfl_xor(rs, 8, 32);
      l_i[r] = l_i[r] * cr + rs;
      m_i[r] = mn;
      corr[r] = cr;
      // D-layout -> row-major P tile in wave-private LDS
      unsigned short* prow = Pw + (r + hf * 8) * 64 + c;
      prow[0]  = f2bf(p0);
      prow[16] = f2bf(p1);
      prow[32] = f2bf(p2);
      prow[48] = f2bf(p3);
    }
    // Rescale running context by exp(m_old - m_new)
#pragma unroll
    for (int j = 0; j < 4; ++j)
#pragma unroll
      for (int r = 0; r < 8; ++r) cacc[j][r] *= corr[r];

    // Intra-wave LDS RAW: wait for P stores before fragment reload
    asm volatile("s_wait_dscnt 0" ::: "memory");
    bf16x16 pa0 = frag_a(Pw, 64, 0, 0);
    bf16x16 pa1 = frag_a(Pw, 64, 0, 32);
#pragma unroll
    for (int j = 0; j < 4; ++j) {
      cacc[j] = WMMA_BF16(pa0, frag_b(Vb, 2048, 16 * j, n0), cacc[j]);
      cacc[j] = WMMA_BF16(pa1, frag_b(Vb, 2048, 16 * j, n0 + 32), cacc[j]);
    }
    // WAR: P reads must finish before next iteration's stores
    asm volatile("s_wait_dscnt 0" ::: "memory");
  }

  // Normalize and store ctx as bf16 [b, t, d] for the output projection
  const int bb = bh >> 4, head = bh & 15;
#pragma unroll
  for (int j = 0; j < 4; ++j)
#pragma unroll
    for (int r = 0; r < 8; ++r) {
      int qrow = q0 + r + hf * 8;
      float v = cacc[j][r] / l_i[r];
      Ctx[((size_t)(bb * 2048 + qrow)) * 1024 + head * 64 + 16 * j + c] = f2bf(v);
    }
}

// ---------------------------------------------------------------------------
// Output projection: out = ctx @ Wo^T + bo   (fp32 output)
// ---------------------------------------------------------------------------
__global__ __launch_bounds__(256) void out_proj_kernel(
    const unsigned short* __restrict__ A, const unsigned short* __restrict__ W,
    const float* __restrict__ bias, float* __restrict__ Out) {
  const int wave = threadIdx.x >> 5;
  const int lane = threadIdx.x & 31;
  const int m0 = blockIdx.x * 256 + wave * 32;
  const int n0 = blockIdx.y * 64;

  f32x8 acc[2][4] = {};
  for (int k = 0; k < 1024; k += 32) {
    if (k + 32 < 1024) {
      __builtin_prefetch(A + (size_t)(m0 + lane) * 1024 + k + 32, 0, 3);
      __builtin_prefetch(W + (size_t)(n0 + lane) * 1024 + k + 32, 0, 3);
      __builtin_prefetch(W + (size_t)(n0 + 32 + lane) * 1024 + k + 32, 0, 3);
    }
    bf16x16 a0 = frag_a(A, 1024, m0, k);
    bf16x16 a1 = frag_a(A, 1024, m0 + 16, k);
#pragma unroll
    for (int j = 0; j < 4; ++j) {
      bf16x16 b = frag_b(W, 1024, n0 + 16 * j, k);
      acc[0][j] = WMMA_BF16(a0, b, acc[0][j]);
      acc[1][j] = WMMA_BF16(a1, b, acc[1][j]);
    }
  }
  const int hf = lane >> 4, c = lane & 15;
#pragma unroll
  for (int i = 0; i < 2; ++i)
#pragma unroll
    for (int j = 0; j < 4; ++j)
#pragma unroll
      for (int r = 0; r < 8; ++r) {
        int gm = m0 + 16 * i + r + hf * 8;
        int gn = n0 + 16 * j + c;
        Out[(size_t)gm * 1024 + gn] = acc[i][j][r] + bias[gn];
      }
}

// ---------------------------------------------------------------------------
// Host orchestration (4 launches total)
// ---------------------------------------------------------------------------
extern "C" void kernel_launch(void* const* d_in, const int* in_sizes, int n_in,
                              void* d_out, int out_size, void* d_ws, size_t ws_size,
                              hipStream_t stream) {
  (void)in_sizes; (void)n_in; (void)out_size; (void)ws_size;

  const float* x  = (const float*)d_in[0];
  const float* Wq = (const float*)d_in[1];
  const float* Wk = (const float*)d_in[2];
  const float* Wv = (const float*)d_in[3];
  const float* Wo = (const float*)d_in[4];
  const float* bo = (const float*)d_in[5];
  float* out = (float*)d_out;

  const int NX = 2 * 2048 * 1024;  // 4,194,304
  const int NW = 1024 * 1024;      // 1,048,576

  // Workspace layout (bf16 elements): 48 MB total
  unsigned short* ws   = (unsigned short*)d_ws;
  unsigned short* xb   = ws;                       // [4096,1024]
  unsigned short* wqb  = xb  + (size_t)NX;         // [1024,1024]
  unsigned short* wkb  = wqb + (size_t)NW;
  unsigned short* wvb  = wkb + (size_t)NW;
  unsigned short* wob  = wvb + (size_t)NW;
  unsigned short* Qb   = wob + (size_t)NW;         // [32,2048,64]
  unsigned short* Kb   = Qb  + (size_t)NX;
  unsigned short* Vtb  = Kb  + (size_t)NX;         // [32,64,2048]
  unsigned short* ctxb = Vtb + (size_t)NX;         // [4096,1024]

  // 1) fp32 -> bf16 for all 5 tensors in a single launch
  CvtArgs ca;
  ca.src[0] = x;  ca.src[1] = Wq; ca.src[2] = Wk; ca.src[3] = Wv; ca.src[4] = Wo;
  ca.dst[0] = (unsigned int*)xb;  ca.dst[1] = (unsigned int*)wqb;
  ca.dst[2] = (unsigned int*)wkb; ca.dst[3] = (unsigned int*)wvb;
  ca.dst[4] = (unsigned int*)wob;
  const int nx8 = NX / 8, nw8 = NW / 8;
  const int ncvt = nx8 + 4 * nw8;
  cvt_all_kernel<<<(ncvt + 255) / 256, 256, 0, stream>>>(ca, nx8, nw8);

  // 2) Fused Q/K/V projections (z selects weight; V stored transposed)
  qkv_proj_kernel<<<dim3(16, 16, 3), 256, 0, stream>>>(xb, wqb, wkb, wvb, Qb, Kb, Vtb);

  // 3) Attention (4096 q-tiles, 8 waves/block)
  attn_kernel<<<512, 256, 0, stream>>>(Qb, Kb, Vtb, ctxb);

  // 4) Output projection + bias (fp32 out)
  out_proj_kernel<<<dim3(16, 16), 256, 0, stream>>>(ctxb, wob, bo, out);
}
